// SimplifiedMambaBlock_69595650064651
// MI455X (gfx1250) — matrix operand order repaired and checked by reading, not compiled, not run on hardware
//
#include <hip/hip_runtime.h>
#include <hip/hip_bf16.h>

// ---------------------------------------------------------------------------
// SimplifiedMambaBlock for MI455X (gfx1250, wave32, WMMA)
// Pipeline:
//   cvt x->f16, transpose-cvt W_in/W_dt/W_out -> f16 (NxK)
//   WMMA GEMM  proj = x @ W_in + b_in                      (8192x2048x1024)
//   conv+SiLU  x_conv (+f16 copy)
//   VALU GEMM  BC = x_conv @ W_bc + b_bc                   (tiny N=32)
//   WMMA GEMM  dt = softplus(x_conv @ W_dt + b_dt)
//   3-phase chunked linear scan (P/S per chunk, combine, replay) + gate -> y_f16
//   WMMA GEMM  out_pre = y @ W_out + b_out
//   residual + LayerNorm -> d_out
// GEMMs use double-buffered LDS tiles filled with GLOBAL_LOAD_ASYNC_TO_LDS_B128
// (ASYNCcnt / s_wait_asynccnt) when the toolchain exposes the builtins.
// ---------------------------------------------------------------------------

typedef __attribute__((ext_vector_type(16))) _Float16 v16h;
typedef __attribute__((ext_vector_type(8)))  _Float16 v8h;
typedef __attribute__((ext_vector_type(8)))  float    v8f;
typedef __attribute__((ext_vector_type(4)))  int      v4i;

// Address-space-qualified pointer types for the async-to-LDS builtin
// (prototype per clang diagnostic: (v4i __device__*, v4i __shared__*, imm, imm))
typedef __attribute__((address_space(1))) v4i* gas_v4i_ptr;
typedef __attribute__((address_space(3))) v4i* las_v4i_ptr;

constexpr int DMODEL  = 1024;
constexpr int DSTATE  = 16;
constexpr int NBATCH  = 4;
constexpr int SEQLEN  = 2048;
constexpr int NROWS   = NBATCH * SEQLEN;      // 8192
constexpr int CHUNK   = 64;
constexpr int NCHUNK  = SEQLEN / CHUNK;       // 32
constexpr float LN_EPS = 1e-5f;

__device__ __forceinline__ float sigmoidf_(float x) { return 1.0f / (1.0f + __expf(-x)); }
__device__ __forceinline__ float siluf_(float x)    { return x * sigmoidf_(x); }
__device__ __forceinline__ float softplusf_(float x){ return (x > 20.0f) ? x : log1pf(__expf(x)); }

#if __has_builtin(__builtin_amdgcn_global_load_async_to_lds_b128) && \
    __has_builtin(__builtin_amdgcn_s_wait_asynccnt)
#define USE_ASYNC_LDS 1
#else
#define USE_ASYNC_LDS 0
#endif

// ---------------------------------------------------------------------------
// f32 -> f16 elementwise
// ---------------------------------------------------------------------------
__global__ void cvt_f16_kernel(const float* __restrict__ in, _Float16* __restrict__ out, int n) {
    int i = blockIdx.x * blockDim.x + threadIdx.x;
    if (i < n) out[i] = (_Float16)in[i];
}

// ---------------------------------------------------------------------------
// Transpose-convert: in (K x N, f32 row-major) -> out (N x K, f16 row-major)
// ---------------------------------------------------------------------------
__global__ void tcvt_kernel(const float* __restrict__ in, _Float16* __restrict__ out,
                            int K, int N) {
    __shared__ float tile[32][33];
    const int nb = blockIdx.x * 32;
    const int kb = blockIdx.y * 32;
    const int tx = threadIdx.x;      // 0..31
    const int ty = threadIdx.y;      // 0..7
    #pragma unroll
    for (int i = 0; i < 32; i += 8)
        tile[ty + i][tx] = in[(size_t)(kb + ty + i) * N + (nb + tx)];
    __syncthreads();
    #pragma unroll
    for (int i = 0; i < 32; i += 8)
        out[(size_t)(nb + ty + i) * K + (kb + tx)] = (_Float16)tile[tx][ty + i];
}

// ---------------------------------------------------------------------------
// WMMA GEMM: C[M,N] = A[M,K](f16) * Bt[N,K](f16)^T + bias, optional softplus.
// Block tile 128(M) x 64(N), BK=32. 256 threads = 8 wave32 waves (4x2),
// each wave computes 32x32 via 2x2 v_wmma_f32_16x16x32_f16.
// Double-buffered LDS; each wave issues exactly 3 async b128 fills per tile
// (768 lane-chunks of 16B = 8KB A + 4KB B), so s_wait_asynccnt(3) drains the
// previous tile while the next stays in flight.
// ---------------------------------------------------------------------------
constexpr int GBM = 128, GBN = 64, GBK = 32, GPAD = 8;   // row stride 40 f16 = 80B (16B aligned)

__device__ __forceinline__ v16h frag16(const _Float16* p0, const _Float16* p1) {
    v8h lo = *(const v8h*)p0;
    v8h hi = *(const v8h*)p1;
    return __builtin_shufflevector(lo, hi, 0,1,2,3,4,5,6,7,8,9,10,11,12,13,14,15);
}

template <int MODE>   // 0: +bias, 1: softplus(+bias)
__global__ __launch_bounds__(256)
void gemm_f16_kernel(const _Float16* __restrict__ A,   // M x K
                     const _Float16* __restrict__ Bt,  // N x K
                     const float* __restrict__ bias,   // N
                     float* __restrict__ C,            // M x N
                     int N, int K) {
    __shared__ __align__(16) _Float16 sA[2][GBM][GBK + GPAD];
    __shared__ __align__(16) _Float16 sB[2][GBN][GBK + GPAD];

    const int tid   = threadIdx.x;
    const int lane  = tid & 31;
    const int wave  = tid >> 5;
    const int waveM = wave >> 1;                 // 0..3
    const int waveN = wave & 1;                  // 0..1
    const int mBlk  = blockIdx.y * GBM;
    const int nBlk  = blockIdx.x * GBN;

    // Issue one block-tile fill: 768 chunks of 16B (A: 512, B: 256), spread as
    // 8 waves x 3 instructions x 32 lanes. Each (wave,j) slice is uniformly
    // all-A or all-B (512 is a multiple of 32).
    auto issue_tile = [&](int buf, int k0) {
        #pragma unroll
        for (int j = 0; j < 3; ++j) {
            const int q = wave * 96 + j * 32 + lane;
            const _Float16* gsrc;
            _Float16* ldst;
            if (q < 512) {
                const int row = q >> 2, cs = q & 3;
                gsrc = A + (size_t)(mBlk + row) * K + k0 + cs * 8;
                ldst = &sA[buf][row][cs * 8];
            } else {
                const int qq = q - 512;
                const int row = qq >> 2, cs = qq & 3;
                gsrc = Bt + (size_t)(nBlk + row) * K + k0 + cs * 8;
                ldst = &sB[buf][row][cs * 8];
            }
#if USE_ASYNC_LDS
            __builtin_amdgcn_global_load_async_to_lds_b128(
                (gas_v4i_ptr)(_Float16*)gsrc,
                (las_v4i_ptr)ldst,
                0, 0);
#else
            __builtin_prefetch(gsrc + GBK, 0, 1);
            *(v8h*)ldst = *(const v8h*)gsrc;
#endif
        }
    };

    v8f acc[2][2] = {};
    const int nK = K / GBK;

    issue_tile(0, 0);
    for (int kt = 0; kt < nK; ++kt) {
        const int cur = kt & 1;
        if (kt + 1 < nK) issue_tile(cur ^ 1, (kt + 1) * GBK);
#if USE_ASYNC_LDS
        if (kt + 1 < nK) __builtin_amdgcn_s_wait_asynccnt(3);  // drain current tile, keep next in flight
        else             __builtin_amdgcn_s_wait_asynccnt(0);
#endif
        __syncthreads();   // all waves' fills for `cur` visible to all

        // A fragment (16x32 f16): lane = row (mod 16); lane-half selects K
        // subgroups {0-7,16-23} vs {8-15,24-31}; 2 f16 per VGPR.
        const int ar = waveM * 32 + (lane & 15);
        const int h8 = (lane >> 4) * 8;
        v16h a0 = frag16(&sA[cur][ar     ][h8], &sA[cur][ar     ][16 + h8]);
        v16h a1 = frag16(&sA[cur][ar + 16][h8], &sA[cur][ar + 16][16 + h8]);

        // B fragment (32x16 f16): lane = col (mod 16); lane-half selects
        // K subgroups {0-7,16-23} vs {8-15,24-31}; 2 f16 per VGPR.
        const int bc = waveN * 32 + (lane & 15);
        const int kh = (lane >> 4) * 8;
        v16h b0 = frag16(&sB[cur][bc     ][kh], &sB[cur][bc     ][kh + 16]);
        v16h b1 = frag16(&sB[cur][bc + 16][kh], &sB[cur][bc + 16][kh + 16]);

        acc[0][0] = __builtin_amdgcn_wmma_f32_16x16x32_f16(false, a0, false, b0, (short)0, acc[0][0], false, false);
        acc[0][1] = __builtin_amdgcn_wmma_f32_16x16x32_f16(false, a0, false, b1, (short)0, acc[0][1], false, false);
        acc[1][0] = __builtin_amdgcn_wmma_f32_16x16x32_f16(false, a1, false, b0, (short)0, acc[1][0], false, false);
        acc[1][1] = __builtin_amdgcn_wmma_f32_16x16x32_f16(false, a1, false, b1, (short)0, acc[1][1], false, false);

        __syncthreads();   // protect buffer cur before next issue overwrites it
    }

    // C/D layout: VGPR e, lanes 0-15 -> M = e, N = lane; lanes 16-31 -> M = 8+e.
    #pragma unroll
    for (int mt = 0; mt < 2; ++mt) {
        #pragma unroll
        for (int nt = 0; nt < 2; ++nt) {
            const int n  = nBlk + waveN * 32 + nt * 16 + (lane & 15);
            const float bv = bias[n];
            const int mb = mBlk + waveM * 32 + mt * 16 + (lane >> 4) * 8;
            #pragma unroll
            for (int e = 0; e < 8; ++e) {
                float v = acc[mt][nt][e] + bv;
                if (MODE == 1) v = softplusf_(v);
                C[(size_t)(mb + e) * N + n] = v;
            }
        }
    }
}

// ---------------------------------------------------------------------------
// Causal depthwise conv (taps=4) + SiLU. proj holds [x_ssm | x_gate] (2D cols).
// Thread = (b,d); blockIdx.y = t-chunk of 256 for parallelism.
// ---------------------------------------------------------------------------
__global__ __launch_bounds__(256)
void conv_silu_kernel(const float* __restrict__ proj,     // NROWS x 2D
                      const float* __restrict__ cw,       // D x 4
                      const float* __restrict__ cb,       // D
                      float* __restrict__ xconv,          // NROWS x D
                      _Float16* __restrict__ xconvh) {    // NROWS x D
    const int g  = blockIdx.x * blockDim.x + threadIdx.x; // 0..4095
    const int b  = g >> 10;
    const int d  = g & (DMODEL - 1);
    const int t0 = blockIdx.y * 256;

    const float w0 = cw[d * 4 + 0], w1 = cw[d * 4 + 1];
    const float w2 = cw[d * 4 + 2], w3 = cw[d * 4 + 3];
    const float bias = cb[d];

    auto ld = [&](int t) -> float {
        return (t >= 0) ? proj[((size_t)(b * SEQLEN + t)) * (2 * DMODEL) + d] : 0.0f;
    };
    float xm3 = ld(t0 - 3), xm2 = ld(t0 - 2), xm1 = ld(t0 - 1);

    for (int tt = 0; tt < 256; ++tt) {
        const int t = t0 + tt;
        const float xc = proj[((size_t)(b * SEQLEN + t)) * (2 * DMODEL) + d];
        const float a  = w0 * xm3 + w1 * xm2 + w2 * xm1 + w3 * xc + bias;
        const float s  = siluf_(a);
        const size_t o = ((size_t)(b * SEQLEN + t)) * DMODEL + d;
        xconv[o]  = s;
        xconvh[o] = (_Float16)s;
        xm3 = xm2; xm2 = xm1; xm1 = xc;
    }
}

// ---------------------------------------------------------------------------
// BC = x_conv @ W_bc + b_bc   (N = 32, VALU: one wave per row, lane = col)
// ---------------------------------------------------------------------------
__global__ __launch_bounds__(256)
void bc_gemm_kernel(const float* __restrict__ xconv,  // NROWS x D
                    const float* __restrict__ Wbc,    // D x 32
                    const float* __restrict__ bbc,    // 32
                    float* __restrict__ BCout) {      // NROWS x 32
    const int lane = threadIdx.x & 31;
    const int warp = threadIdx.x >> 5;
    const size_t row = (size_t)blockIdx.x * 8 + warp;
    const float* xr = xconv + row * DMODEL;
    float acc = 0.0f;
    #pragma unroll 4
    for (int k = 0; k < DMODEL; ++k)
        acc = fmaf(xr[k], Wbc[k * 32 + lane], acc);
    BCout[row * 32 + lane] = acc + bbc[lane];
}

// ---------------------------------------------------------------------------
// Scan phase A: per (b,chunk,d) compute cumulative decay P and local state S
// over the chunk assuming h0 = 0.
// ---------------------------------------------------------------------------
__global__ __launch_bounds__(1024)
void scanA_kernel(const float* __restrict__ dt,     // NROWS x D  (post-softplus)
                  const float* __restrict__ xconv,  // NROWS x D
                  const float* __restrict__ BCbuf,  // NROWS x 32 (B | C)
                  const float* __restrict__ A_log,  // D x 16
                  float* __restrict__ Pbuf,         // B x NCHUNK x D x 16
                  float* __restrict__ Sbuf) {
    const int d = threadIdx.x;
    const int c = blockIdx.x;
    const int b = blockIdx.y;
    const int t0 = c * CHUNK;

    __shared__ float sBm[CHUNK][DSTATE];
    {
        const int tt = d >> 4, n = d & 15;
        sBm[tt][n] = BCbuf[((size_t)(b * SEQLEN + t0 + tt)) * 32 + n];
    }
    __syncthreads();

    float a[DSTATE];
    #pragma unroll
    for (int n = 0; n < DSTATE; ++n) a[n] = -__expf(A_log[d * DSTATE + n]);

    float p[DSTATE], s[DSTATE];
    #pragma unroll
    for (int n = 0; n < DSTATE; ++n) { p[n] = 1.0f; s[n] = 0.0f; }

    for (int tt = 0; tt < CHUNK; ++tt) {
        const size_t idx = ((size_t)(b * SEQLEN + t0 + tt)) * DMODEL + d;
        const float dtt = dt[idx];
        const float bx  = dtt * xconv[idx];
        #pragma unroll
        for (int n = 0; n < DSTATE; ++n) {
            const float dA = __expf(dtt * a[n]);
            p[n] *= dA;
            s[n] = s[n] * dA + bx * sBm[tt][n];
        }
    }

    const size_t o = (((size_t)b * NCHUNK + c) * DMODEL + d) * DSTATE;
    #pragma unroll
    for (int q = 0; q < 4; ++q) {
        *(float4*)&Pbuf[o + q * 4] = make_float4(p[q*4], p[q*4+1], p[q*4+2], p[q*4+3]);
        *(float4*)&Sbuf[o + q * 4] = make_float4(s[q*4], s[q*4+1], s[q*4+2], s[q*4+3]);
    }
}

// ---------------------------------------------------------------------------
// Scan phase B: sequential combine across chunks (32 steps); emit carry-in H
// for each chunk: Hin[c] = H before chunk c; H = H*P_c + S_c.
// ---------------------------------------------------------------------------
__global__ __launch_bounds__(1024)
void scanB_kernel(const float* __restrict__ Pbuf, const float* __restrict__ Sbuf,
                  float* __restrict__ Hin) {
    const int d = threadIdx.x;
    const int b = blockIdx.x;
    float h[DSTATE];
    #pragma unroll
    for (int n = 0; n < DSTATE; ++n) h[n] = 0.0f;
    for (int c = 0; c < NCHUNK; ++c) {
        const size_t o = (((size_t)b * NCHUNK + c) * DMODEL + d) * DSTATE;
        #pragma unroll
        for (int n = 0; n < DSTATE; ++n) Hin[o + n] = h[n];
        #pragma unroll
        for (int n = 0; n < DSTATE; ++n) h[n] = h[n] * Pbuf[o + n] + Sbuf[o + n];
    }
}

// ---------------------------------------------------------------------------
// Scan phase C: replay each chunk with its carry-in state; y = h.C + x*D,
// fused with SiLU gate; emit y_gated as f16 for the output GEMM.
// ---------------------------------------------------------------------------
__global__ __launch_bounds__(1024)
void scanC_kernel(const float* __restrict__ dt,
                  const float* __restrict__ xconv,
                  const float* __restrict__ BCbuf,
                  const float* __restrict__ A_log,
                  const float* __restrict__ Dp,     // D
                  const float* __restrict__ Hin,
                  const float* __restrict__ proj,   // for x_gate (cols D..2D)
                  _Float16* __restrict__ yg) {      // NROWS x D
    const int d = threadIdx.x;
    const int c = blockIdx.x;
    const int b = blockIdx.y;
    const int t0 = c * CHUNK;

    __shared__ float sBm[CHUNK][DSTATE];
    __shared__ float sCm[CHUNK][DSTATE];
    {
        const int tt = d >> 4, n = d & 15;
        const size_t r = ((size_t)(b * SEQLEN + t0 + tt)) * 32;
        sBm[tt][n] = BCbuf[r + n];
        sCm[tt][n] = BCbuf[r + 16 + n];
    }
    __syncthreads();

    float a[DSTATE];
    #pragma unroll
    for (int n = 0; n < DSTATE; ++n) a[n] = -__expf(A_log[d * DSTATE + n]);
    const float Dd = Dp[d];

    float h[DSTATE];
    {
        const size_t o = (((size_t)b * NCHUNK + c) * DMODEL + d) * DSTATE;
        #pragma unroll
        for (int n = 0; n < DSTATE; ++n) h[n] = Hin[o + n];
    }

    for (int tt = 0; tt < CHUNK; ++tt) {
        const size_t row = (size_t)(b * SEQLEN + t0 + tt);
        const size_t idx = row * DMODEL + d;
        const float dtt = dt[idx];
        const float xt  = xconv[idx];
        const float bx  = dtt * xt;
        float y = 0.0f;
        #pragma unroll
        for (int n = 0; n < DSTATE; ++n) {
            const float dA = __expf(dtt * a[n]);
            h[n] = h[n] * dA + bx * sBm[tt][n];
            y = fmaf(h[n], sCm[tt][n], y);
        }
        y = fmaf(xt, Dd, y);
        const float g = proj[row * (2 * DMODEL) + DMODEL + d];
        yg[idx] = (_Float16)(y * siluf_(g));
    }
}

// ---------------------------------------------------------------------------
// Residual + LayerNorm: out = LN(out_pre + x) * g + b, one block per row.
// ---------------------------------------------------------------------------
__global__ __launch_bounds__(256)
void ln_kernel(const float* __restrict__ outpre, const float* __restrict__ x,
               const float* __restrict__ g, const float* __restrict__ bb,
               float* __restrict__ out) {
    const size_t row = blockIdx.x;
    const int tid = threadIdx.x;
    __shared__ float rsum[256], rsq[256];

    float v[4], sum = 0.0f, sq = 0.0f;
    #pragma unroll
    for (int i = 0; i < 4; ++i) {
        const int col = tid + i * 256;
        const float t = outpre[row * DMODEL + col] + x[row * DMODEL + col];
        v[i] = t; sum += t; sq += t * t;
    }
    rsum[tid] = sum; rsq[tid] = sq;
    __syncthreads();
    for (int s = 128; s > 0; s >>= 1) {
        if (tid < s) { rsum[tid] += rsum[tid + s]; rsq[tid] += rsq[tid + s]; }
        __syncthreads();
    }
    const float mu   = rsum[0] * (1.0f / DMODEL);
    const float var  = rsq[0] * (1.0f / DMODEL) - mu * mu;
    const float rstd = rsqrtf(var + LN_EPS);
    #pragma unroll
    for (int i = 0; i < 4; ++i) {
        const int col = tid + i * 256;
        out[row * DMODEL + col] = (v[i] - mu) * rstd * g[col] + bb[col];
    }
}

// ---------------------------------------------------------------------------
// Launcher
// ---------------------------------------------------------------------------
extern "C" void kernel_launch(void* const* d_in, const int* in_sizes, int n_in,
                              void* d_out, int out_size, void* d_ws, size_t ws_size,
                              hipStream_t stream) {
    const float* x      = (const float*)d_in[0];
    const float* W_in   = (const float*)d_in[1];
    const float* b_in   = (const float*)d_in[2];
    const float* conv_w = (const float*)d_in[3];
    const float* conv_b = (const float*)d_in[4];
    const float* W_bc   = (const float*)d_in[5];
    const float* b_bc   = (const float*)d_in[6];
    const float* W_dt   = (const float*)d_in[7];
    const float* b_dt   = (const float*)d_in[8];
    const float* A_log  = (const float*)d_in[9];
    const float* Dp     = (const float*)d_in[10];
    const float* W_out  = (const float*)d_in[11];
    const float* b_out  = (const float*)d_in[12];
    const float* ln_g   = (const float*)d_in[13];
    const float* ln_b   = (const float*)d_in[14];
    float* out = (float*)d_out;

    // Workspace carve-up (256B aligned)
    char* ws = (char*)d_ws;
    size_t off = 0;
    auto carve = [&](size_t bytes) -> char* {
        char* p = ws + off;
        off += (bytes + 255) & ~(size_t)255;
        return p;
    };
    _Float16* xf16    = (_Float16*)carve((size_t)NROWS * DMODEL * 2);
    _Float16* WinT    = (_Float16*)carve((size_t)2 * DMODEL * DMODEL * 2);
    _Float16* WdtT    = (_Float16*)carve((size_t)DMODEL * DMODEL * 2);
    _Float16* WoutT   = (_Float16*)carve((size_t)DMODEL * DMODEL * 2);
    float*    proj    = (float*)   carve((size_t)NROWS * 2 * DMODEL * 4);
    float*    xconv   = (float*)   carve((size_t)NROWS * DMODEL * 4);
    _Float16* xconvh  = (_Float16*)carve((size_t)NROWS * DMODEL * 2);
    float*    BCbuf   = (float*)   carve((size_t)NROWS * 32 * 4);
    float*    dtbuf   = (float*)   carve((size_t)NROWS * DMODEL * 4);
    float*    Pbuf    = (float*)   carve((size_t)NBATCH * NCHUNK * DMODEL * DSTATE * 4);
    float*    Sbuf    = (float*)   carve((size_t)NBATCH * NCHUNK * DMODEL * DSTATE * 4);
    float*    Hin     = (float*)   carve((size_t)NBATCH * NCHUNK * DMODEL * DSTATE * 4);
    _Float16* ygf16   = (_Float16*)carve((size_t)NROWS * DMODEL * 2);
    float*    outpre  = (float*)   carve((size_t)NROWS * DMODEL * 4);
    (void)ws_size; (void)in_sizes; (void)n_in; (void)out_size;

    // 1) precision conversion
    {
        const int n = NROWS * DMODEL;
        cvt_f16_kernel<<<(n + 255) / 256, 256, 0, stream>>>(x, xf16, n);
        tcvt_kernel<<<dim3(2 * DMODEL / 32, DMODEL / 32), dim3(32, 8), 0, stream>>>(W_in,  WinT,  DMODEL, 2 * DMODEL);
        tcvt_kernel<<<dim3(DMODEL / 32,     DMODEL / 32), dim3(32, 8), 0, stream>>>(W_dt,  WdtT,  DMODEL, DMODEL);
        tcvt_kernel<<<dim3(DMODEL / 32,     DMODEL / 32), dim3(32, 8), 0, stream>>>(W_out, WoutT, DMODEL, DMODEL);
    }

    // 2) proj = x @ W_in + b_in   (M=8192, N=2048, K=1024)
    gemm_f16_kernel<0><<<dim3(2 * DMODEL / GBN, NROWS / GBM), 256, 0, stream>>>(
        xf16, WinT, b_in, proj, 2 * DMODEL, DMODEL);

    // 3) depthwise conv + SiLU
    conv_silu_kernel<<<dim3(NBATCH * DMODEL / 256, SEQLEN / 256), 256, 0, stream>>>(
        proj, conv_w, conv_b, xconv, xconvh);

    // 4) BC = x_conv @ W_bc + b_bc
    bc_gemm_kernel<<<NROWS / 8, 256, 0, stream>>>(xconv, W_bc, b_bc, BCbuf);

    // 5) dt = softplus(x_conv @ W_dt + b_dt)
    gemm_f16_kernel<1><<<dim3(DMODEL / GBN, NROWS / GBM), 256, 0, stream>>>(
        xconvh, WdtT, b_dt, dtbuf, DMODEL, DMODEL);

    // 6) chunked linear scan + gate
    scanA_kernel<<<dim3(NCHUNK, NBATCH), DMODEL, 0, stream>>>(dtbuf, xconv, BCbuf, A_log, Pbuf, Sbuf);
    scanB_kernel<<<NBATCH, DMODEL, 0, stream>>>(Pbuf, Sbuf, Hin);
    scanC_kernel<<<dim3(NCHUNK, NBATCH), DMODEL, 0, stream>>>(dtbuf, xconv, BCbuf, A_log, Dp, Hin, proj, ygf16);

    // 7) out_pre = y_gated @ W_out + b_out
    gemm_f16_kernel<0><<<dim3(DMODEL / GBN, NROWS / GBM), 256, 0, stream>>>(
        ygf16, WoutT, b_out, outpre, DMODEL, DMODEL);

    // 8) residual + LayerNorm
    ln_kernel<<<NROWS, 256, 0, stream>>>(outpre, x, ln_g, ln_b, out);
}
